// VectorQuantization_30966714204292
// MI455X (gfx1250) — compile-verified, matrix operand order checked by reading, not run on hardware
//
#include <hip/hip_runtime.h>
#include <hip/hip_bf16.h>
#include <math.h>

// ---------------------------------------------------------------------------
// CDNA5 (gfx1250) fused vector-quantization kernel.
//   quantized = softmax(-dist + gumbel(u)) @ codebook
//   ind       = argmin(dist)
//   loss      = (1+BETA) * sum((x - quantized)^2)
// Flash-softmax structure, bf16 WMMA (v_wmma_f32_16x16x32_bf16) for both
// GEMMs, two TDM tensor_load_to_lds per tile (row-major + transposed
// codebook) so every WMMA operand is a pair of ds_load_b128.
// ---------------------------------------------------------------------------

typedef __attribute__((ext_vector_type(16))) __bf16        v16bf;
typedef __attribute__((ext_vector_type(8)))  __bf16        v8bf;
typedef __attribute__((ext_vector_type(8)))  float         v8f;
typedef __attribute__((ext_vector_type(4))) unsigned int   u32x4;
typedef __attribute__((ext_vector_type(8)))  int           i32x8;
typedef __attribute__((ext_vector_type(4)))  int           i32x4;

#define VQ_EPS  1e-20f
#define VQ_BETA 0.25f

#define BDIM 8192
#define NDIM 8192
#define DDIM 512

#define MT        16        // x rows per workgroup
#define NT        128       // codebook rows per tile iteration
#define NWAVES    8
#define NTHREADS  256
#define CB_STRIDE  520      // [n][d] tile row stride (512 + 8 pad bf16)
#define CBT_STRIDE 136      // [d][n] tile row stride (128 + 8 pad bf16)
#define XS_STRIDE  520
#define P_STRIDE   136

// ---- LDS layout (bytes), dynamic shared; every tile base 16B aligned ----
#define OFF_CB   0
#define SZ_CB    (NT * CB_STRIDE * 2)           // 133120
#define OFF_CBT  (OFF_CB + SZ_CB)               // 133120
#define SZ_CBT   (DDIM * CBT_STRIDE * 2)        // 139264
#define OFF_XS   (OFF_CBT + SZ_CBT)             // 272384
#define SZ_XS    (MT * XS_STRIDE * 2)           // 16640
#define OFF_S    (OFF_XS + SZ_XS)               // 289024
#define SZ_S     (MT * NT * 4)                  // 8192
#define OFF_P    (OFF_S + SZ_S)                 // 297216
#define SZ_P     (MT * P_STRIDE * 2)            // 4352
#define OFF_R1   (OFF_P + SZ_P)                 // 301568
#define OFF_R2   (OFF_R1 + NTHREADS * 4)
#define OFF_RI   (OFF_R2 + NTHREADS * 4)
#define OFF_XX   (OFF_RI + NTHREADS * 4)
#define OFF_RM   (OFF_XX + MT * 4)
#define OFF_RL   (OFF_RM + MT * 4)
#define OFF_RA   (OFF_RL + MT * 4)
#define OFF_RD   (OFF_RA + MT * 4)
#define OFF_RDI  (OFF_RD + MT * 4)
#define OFF_LP   (OFF_RDI + MT * 4)
#define SZ_LP    (MT * NWAVES * 4)
#define SMEM_BYTES (OFF_LP + SZ_LP)             // 305536 B < 320 KB WGP LDS

#if defined(__has_builtin)
#if __has_builtin(__builtin_amdgcn_tensor_load_to_lds) && \
    __has_builtin(__builtin_amdgcn_s_wait_tensorcnt)
#define USE_TDM 1
#endif
#endif

__device__ __forceinline__ v8f wmma_bf16(v16bf a, v16bf b, v8f c) {
  // D = A(16x32 bf16) * B(32x16 bf16) + C(16x16 f32)
  return __builtin_amdgcn_wmma_f32_16x16x32_bf16(
      false, a, false, b, (short)0, c, false, false);
}

// Row-major fragment load (A operand, or B operand whose K dim is contiguous
// in the LDS row).  ISA 7.12.2 16-bit operand layout:
//   lanes 0-15 : row = lane,    K slots {0..7, 16..23}
//   lanes 16-31: row = lane-16, K slots {8..15, 24..31}
// Per lane this is two contiguous 16B runs -> two ds_load_b128.
__device__ __forceinline__ v16bf load_frag(const __bf16* lds, int stride,
                                           int row_base, int k_base, int lane) {
  int r  = row_base + (lane & 15);
  int kb = (lane >> 4) << 3;  // 0 or 8
  const __bf16* p = lds + r * stride + k_base + kb;
  v8bf lo = *(const v8bf*)(p);       // slots kb .. kb+7
  v8bf hi = *(const v8bf*)(p + 16);  // slots kb+16 .. kb+23
  return __builtin_shufflevector(lo, hi, 0, 1, 2, 3, 4, 5, 6, 7,
                                 8, 9, 10, 11, 12, 13, 14, 15);
}

// ---------------------------------------------------------------------------
// Prep: cnorm[n] = ||codebook[n]||^2, plus bf16 copies of the codebook in
// row-major [N][D] and transposed [D][N] (both live in the 192MB L2 after
// first touch; 16 MB total).
// ---------------------------------------------------------------------------
__global__ __launch_bounds__(256) void vq_prep(const float* __restrict__ cb,
                                               unsigned short* __restrict__ cb16_raw,
                                               unsigned short* __restrict__ cbt16_raw,
                                               float* __restrict__ cnorm) {
  __shared__ float red[256];
  __bf16* cb16  = (__bf16*)cb16_raw;
  __bf16* cbt16 = (__bf16*)cbt16_raw;
  int t   = threadIdx.x;
  int row = blockIdx.x * 16 + (t >> 4);
  float s = 0.f;
#pragma unroll 4
  for (int j = 0; j < DDIM / 16; ++j) {
    int d   = (t & 15) + j * 16;
    float v = cb[(size_t)row * DDIM + d];
    __bf16 bv = (__bf16)v;
    cb16[(size_t)row * DDIM + d]  = bv;
    cbt16[(size_t)d * NDIM + row] = bv;
    s += v * v;
  }
  red[t] = s;
  __syncthreads();
  if ((t & 15) == 0) {
    float acc = 0.f;
#pragma unroll
    for (int j = 0; j < 16; ++j) acc += red[(t & ~15) + j];
    cnorm[row] = acc;
  }
}

// ---------------------------------------------------------------------------
// Main fused kernel: one block = 16 rows of x, 8 waves.
// ---------------------------------------------------------------------------
__global__ __launch_bounds__(NTHREADS) void vq_main(
    const float* __restrict__ x, const float* __restrict__ u,
    const unsigned short* __restrict__ cb16_raw,
    const unsigned short* __restrict__ cbt16_raw,
    const float* __restrict__ cnorm, float* __restrict__ out) {
  extern __shared__ char smem[];
  __bf16* cbT  = (__bf16*)(smem + OFF_CB);   // [n][d] tile
  __bf16* cbtT = (__bf16*)(smem + OFF_CBT);  // [d][n] tile
  __bf16* xsT  = (__bf16*)(smem + OFF_XS);
  float*  sT   = (float*)(smem + OFF_S);
  __bf16* pT   = (__bf16*)(smem + OFF_P);
  float*  red1 = (float*)(smem + OFF_R1);
  float*  red2 = (float*)(smem + OFF_R2);
  int*    redi = (int*)(smem + OFF_RI);
  float*  xx   = (float*)(smem + OFF_XX);
  float*  rowM = (float*)(smem + OFF_RM);
  float*  rowL = (float*)(smem + OFF_RL);
  float*  rowA = (float*)(smem + OFF_RA);
  float*  rowD = (float*)(smem + OFF_RD);
  int*    rowI = (int*)(smem + OFF_RDI);
  float*  lossP = (float*)(smem + OFF_LP);

  const __bf16* cb16  = (const __bf16*)cb16_raw;
  const __bf16* cbt16 = (const __bf16*)cbt16_raw;
  const int t    = threadIdx.x;
  const int wave = t >> 5;
  const int lane = t & 31;
  const int row0 = blockIdx.x * MT;

  // ---- init: x tile -> bf16 LDS, ||x||^2, softmax/argmin state ----
  {
    int m = t >> 4;
    float s = 0.f;
#pragma unroll 4
    for (int j = 0; j < DDIM / 16; ++j) {
      int d   = (t & 15) + j * 16;
      float v = x[(size_t)(row0 + m) * DDIM + d];
      xsT[m * XS_STRIDE + d] = (__bf16)v;
      s += v * v;
    }
    red1[t] = s;
  }
  __syncthreads();
  if (t < MT) {
    float s = 0.f;
#pragma unroll
    for (int j = 0; j < 16; ++j) s += red1[t * 16 + j];
    xx[t]   = s;
    rowM[t] = -INFINITY;
    rowL[t] = 0.f;
    rowD[t] = INFINITY;
    rowI[t] = 0;
  }

  // Per-wave output accumulator: D-slice of 64 cols = 4 WMMA C tiles.
  v8f O[4];
#pragma unroll
  for (int j = 0; j < 4; ++j)
#pragma unroll
    for (int i = 0; i < 8; ++i) O[j][i] = 0.f;

  for (int n0 = 0; n0 < NDIM; n0 += NT) {
    __syncthreads();  // previous GEMM2 done reading tiles

    // ---- stage codebook tiles into padded LDS via TDM ----
#if defined(USE_TDM)
    if (wave == 0) {
      // Tile A: rows n0..n0+127 of [N][D] bf16, row pad 16B -> stride 520.
      unsigned long long ga =
          (unsigned long long)(uintptr_t)cb16 + (unsigned long long)n0 * (DDIM * 2);
      u32x4 g0;
      g0[0] = 1u;                                   // count=1
      g0[1] = (unsigned int)(uintptr_t)(smem + OFF_CB);
      g0[2] = (unsigned int)ga;
      g0[3] = (unsigned int)((ga >> 32) & 0x1FFFFFFu) | (2u << 30);  // type=2
      i32x8 g1;
      // data_size=1(2B), pad_enable, pad_interval=7(256DW=one 512-elem row),
      // pad_amount=3(4DW=16B)
      g1[0] = (1 << 16) | (1 << 20) | (7 << 22) | (3 << 25);
      g1[1] = (DDIM & 0xFFFF) << 16;  // tensor_dim0 = 512
      g1[2] = (NDIM & 0xFFFF) << 16;  // tensor_dim1 = 8192
      g1[3] = (DDIM & 0xFFFF) << 16;  // tile_dim0 = 512
      g1[4] = NT;                     // tile_dim1 = 128
      g1[5] = DDIM;                   // tensor_dim0_stride = 512
      g1[6] = 0;
      g1[7] = 0;
      i32x4 z4 = {0, 0, 0, 0};
      i32x8 z8 = {0, 0, 0, 0, 0, 0, 0, 0};
      __builtin_amdgcn_tensor_load_to_lds(g0, g1, z4, z4, z8, 0);
      __builtin_amdgcn_s_wait_tensorcnt(0);
    } else if (wave == 1) {
      // Tile B: cols n0..n0+127 of transposed [D][N] bf16, stride 136.
      unsigned long long ga =
          (unsigned long long)(uintptr_t)cbt16 + (unsigned long long)n0 * 2;
      u32x4 g0;
      g0[0] = 1u;
      g0[1] = (unsigned int)(uintptr_t)(smem + OFF_CBT);
      g0[2] = (unsigned int)ga;
      g0[3] = (unsigned int)((ga >> 32) & 0x1FFFFFFu) | (2u << 30);
      i32x8 g1;
      // pad_interval=5(64DW=one 128-elem row), pad_amount=3(4DW=16B)
      g1[0] = (1 << 16) | (1 << 20) | (5 << 22) | (3 << 25);
      g1[1] = (NDIM & 0xFFFF) << 16;  // tensor_dim0 = 8192
      g1[2] = (DDIM & 0xFFFF) << 16;  // tensor_dim1 = 512
      g1[3] = (NT & 0xFFFF) << 16;    // tile_dim0 = 128
      g1[4] = DDIM;                   // tile_dim1 = 512
      g1[5] = NDIM;                   // tensor_dim0_stride = 8192
      g1[6] = 0;
      g1[7] = 0;
      i32x4 z4 = {0, 0, 0, 0};
      i32x8 z8 = {0, 0, 0, 0, 0, 0, 0, 0};
      __builtin_amdgcn_tensor_load_to_lds(g0, g1, z4, z4, z8, 0);
      __builtin_amdgcn_s_wait_tensorcnt(0);
    }
#else
    {
      const uint4* src = (const uint4*)(cb16 + (size_t)n0 * DDIM);
#pragma unroll 4
      for (int c = t; c < NT * (DDIM / 8); c += NTHREADS) {
        int r = c >> 6, db = c & 63;  // 64 16B-chunks per 512-elem row
        uint4 v = src[r * (DDIM / 8) + db];
        *(uint4*)((char*)cbT + (r * CB_STRIDE + db * 8) * 2) = v;
      }
#pragma unroll 4
      for (int c = t; c < DDIM * (NT / 8); c += NTHREADS) {
        int r = c >> 4, db = c & 15;  // 16 16B-chunks per 128-elem row
        uint4 v = *(const uint4*)(cbt16 + (size_t)r * NDIM + n0 + db * 8);
        *(uint4*)((char*)cbtT + (r * CBT_STRIDE + db * 8) * 2) = v;
      }
    }
#endif
    __syncthreads();

    // ---- GEMM1: S[16 x NT] = Xtile * Ctile^T ; wave owns one 16-N subtile --
    {
      v8f s;
#pragma unroll
      for (int i = 0; i < 8; ++i) s[i] = 0.f;
      int nb = wave * 16;
#pragma unroll
      for (int kt = 0; kt < DDIM / 32; ++kt) {
        v16bf a = load_frag(xsT, XS_STRIDE, 0, kt * 32, lane);   // A: x rows
        v16bf b = load_frag(cbT, CB_STRIDE, nb, kt * 32, lane);  // B: cb rows
        s = wmma_bf16(a, b, s);
      }
      int mhi = (lane >> 4) * 8;
      int n   = nb + (lane & 15);
#pragma unroll
      for (int i = 0; i < 8; ++i) sT[(mhi + i) * NT + n] = s[i];
    }
    __syncthreads();

    // ---- online softmax pass 1: gumbel + dist + logits + local max/argmin --
    {
      int m = t >> 4, c0 = t & 15;
      float xxm = xx[m];
      float lmax = -INFINITY, dmn = INFINITY;
      int dix = 0;
#pragma unroll
      for (int j = 0; j < NT / 16; ++j) {
        int n    = c0 + j * 16;
        float s  = sT[m * NT + n];
        float cn = cnorm[n0 + n];
        float uv = u[(size_t)(row0 + m) * NDIM + n0 + n];
        float g  = -logf(-logf(uv + VQ_EPS) + VQ_EPS);
        float dist = xxm + cn - 2.0f * s;
        float lg   = g - dist;
        sT[m * NT + n] = lg;  // overwrite scores with logits
        lmax = fmaxf(lmax, lg);
        if (dist < dmn) { dmn = dist; dix = n0 + n; }
      }
      red1[t] = lmax;
      red2[t] = dmn;
      redi[t] = dix;
    }
    __syncthreads();
    if (t < MT) {
      float tm = -INFINITY, dm = rowD[t];
      int di = rowI[t];
#pragma unroll
      for (int j = 0; j < 16; ++j) {
        tm = fmaxf(tm, red1[t * 16 + j]);
        float dv = red2[t * 16 + j];
        int ix = redi[t * 16 + j];
        if (dv < dm || (dv == dm && ix < di)) { dm = dv; di = ix; }
      }
      rowD[t] = dm;
      rowI[t] = di;
      float mo = rowM[t];
      float mn = fmaxf(mo, tm);
      rowM[t] = mn;
      rowA[t] = expf(mo - mn);  // alpha rescale factor
    }
    __syncthreads();

    // ---- pass 2: P = exp(logit - m_new), bf16 to LDS, partial row sums ----
    {
      int m = t >> 4, c0 = t & 15;
      float mn = rowM[m];
      float ps = 0.f;
#pragma unroll
      for (int j = 0; j < NT / 16; ++j) {
        int n = c0 + j * 16;
        float p = expf(sT[m * NT + n] - mn);
        ps += p;
        pT[m * P_STRIDE + n] = (__bf16)p;
      }
      red1[t] = ps;
    }
    __syncthreads();
    if (t < MT) {
      float s = 0.f;
#pragma unroll
      for (int j = 0; j < 16; ++j) s += red1[t * 16 + j];
      rowL[t] = rowL[t] * rowA[t] + s;
    }
    __syncthreads();

    // ---- GEMM2: O[16 x 64] = alpha*O + P(16xNT) * Ctile(NT x 64-slice) ----
    {
      int mhi = (lane >> 4) * 8;
      float al[8];
#pragma unroll
      for (int i = 0; i < 8; ++i) al[i] = rowA[mhi + i];
#pragma unroll
      for (int j = 0; j < 4; ++j)
#pragma unroll
        for (int i = 0; i < 8; ++i) O[j][i] *= al[i];

      v16bf pa[NT / 32];
#pragma unroll
      for (int kt = 0; kt < NT / 32; ++kt)
        pa[kt] = load_frag(pT, P_STRIDE, 0, kt * 32, lane);  // A: P rows

      int dbase = wave * 64;
#pragma unroll
      for (int j = 0; j < 4; ++j) {
#pragma unroll
        for (int kt = 0; kt < NT / 32; ++kt) {
          // B[k=n][col=d] from transposed tile: row=d, contiguous n.
          v16bf b = load_frag(cbtT, CBT_STRIDE, dbase + j * 16, kt * 32, lane);
          O[j] = wmma_bf16(pa[kt], b, O[j]);
        }
      }
    }
  }

  // ---- epilogue: normalize, store quantized, loss reduction, ind ----
  __syncthreads();
  if (t < MT) rowA[t] = 1.0f / rowL[t];  // reuse as inv row sum
  __syncthreads();
  {
    int mhi   = (lane >> 4) * 8;
    int dbase = wave * 64;
    float invl[8];
#pragma unroll
    for (int i = 0; i < 8; ++i) invl[i] = rowA[mhi + i];
    float lp[8];
#pragma unroll
    for (int i = 0; i < 8; ++i) lp[i] = 0.f;

#pragma unroll
    for (int j = 0; j < 4; ++j) {
      int d = dbase + j * 16 + (lane & 15);
#pragma unroll
      for (int i = 0; i < 8; ++i) {
        int m = mhi + i;
        float q = O[j][i] * invl[i];
        out[(size_t)(row0 + m) * DDIM + d] = q;
        float xv = x[(size_t)(row0 + m) * DDIM + d];
        float df = xv - q;
        lp[i] += df * df;
      }
    }
    // reduce over the 16 lanes that share each m
#pragma unroll
    for (int i = 0; i < 8; ++i) {
      float v = lp[i];
      for (int off = 8; off; off >>= 1) v += __shfl_xor(v, off, 16);
      if ((lane & 15) == 0) lossP[(mhi + i) * NWAVES + wave] = v;
    }
  }
  __syncthreads();
  if (t < MT) {
    float acc = 0.f;
#pragma unroll
    for (int w = 0; w < NWAVES; ++w) acc += lossP[t * NWAVES + w];
    out[(size_t)BDIM * DDIM + row0 + t]        = (float)rowI[t];
    out[(size_t)BDIM * DDIM + BDIM + row0 + t] = (1.0f + VQ_BETA) * acc;
  }
}

// ---------------------------------------------------------------------------
extern "C" void kernel_launch(void* const* d_in, const int* in_sizes, int n_in,
                              void* d_out, int out_size, void* d_ws,
                              size_t ws_size, hipStream_t stream) {
  (void)in_sizes; (void)n_in; (void)out_size; (void)ws_size;
  const float* x  = (const float*)d_in[0];
  const float* cb = (const float*)d_in[1];
  const float* u  = (const float*)d_in[2];
  float* out      = (float*)d_out;

  unsigned short* cb16  = (unsigned short*)d_ws;                          // 8 MB
  unsigned short* cbt16 = cb16 + (size_t)NDIM * DDIM;                     // 8 MB
  float* cnorm = (float*)((char*)d_ws + (size_t)NDIM * DDIM * 4);         // 32 KB

  vq_prep<<<NDIM / 16, 256, 0, stream>>>(cb, cb16, cbt16, cnorm);
  vq_main<<<BDIM / MT, NTHREADS, SMEM_BYTES, stream>>>(x, u, cb16, cbt16, cnorm, out);
}
